// GraphAttentionLayer_46772193853548
// MI455X (gfx1250) — compile-verified, matrix operand order checked by reading, not run on hardware
//
#include <hip/hip_runtime.h>
#include <hip/hip_bf16.h>

// GAT layer, MI455X (gfx1250, wave32, WMMA).
//
// Math: softmax_j(src_i + dst_j) == softmax_j(dst_j)  (src_i cancels), so
//   alpha[h,i,j] = adj[i,j]*E_h[j] / (adj[i,:] . E_h),  E_h[j]=exp(dst[h,j]-max_h)
//   out = (1/8) sum_h diag(1/S_h) * (A @ (E_h*h)),  S_h = A @ E_h
// Everything folds into ONE f16 WMMA GEMM:  T = A(4096x4096) @ P(4096x528),
// P cols [h*64+f] = E_h[j]*h[j,f], cols 512..519 = E_h[j], pad to 528.
//
// The GEMM is compute bound (~17.8 GFLOP of v_wmma_f32_16x16x32_f16 vs a
// 64MB/23.3TB/s = 2.7us HBM floor). Hot loop: two-deep software pipeline,
// two named register sets (no rotation movs), pure b128 loads + v_wmma,
// no LDS, no barriers; adjacency pre-converted once to f16 fragment layout.

typedef __attribute__((ext_vector_type(16))) _Float16 v16h;
typedef __attribute__((ext_vector_type(2)))  _Float16 v2h;
typedef __attribute__((ext_vector_type(8)))  float    v8f;

#define NNODES 4096
#define IN_F   256
#define OUT_F  64
#define NHEADS 8
#define NPAD   528      // 520 used cols padded to 33 tiles of 16
#define NTILES 33
#define KSTEPS (NNODES / 32)
#define MTILES (NNODES / 16)

// ---- WMMA 16-bit fragment index helpers (ISA 7.12.2) --------------------
// A 16x32 f16 (mirrored for B 32x16): lane = (row&15) | ((koff>>3&1)<<4);
// element e <-> koff:  koff = ((e&8)<<1) + ((lane>>4)<<3) + (e&7)
__device__ __forceinline__ int frag_lane(int rc, int koff) {
    return (rc & 15) | (((koff >> 3) & 1) << 4);
}
__device__ __forceinline__ int frag_e(int koff) {
    return ((koff >> 4) << 3) | (koff & 7);
}

__device__ __forceinline__ v8f wmma_f16(v16h a, v16h b, v8f c) {
    return __builtin_amdgcn_wmma_f32_16x16x32_f16(false, a, false, b,
                                                  (short)0, c, false, false);
}

// ======================= kernel 1: H = X @ W (WMMA) ======================
__global__ __launch_bounds__(256) void k_feat_gemm(
    const float* __restrict__ X, const float* __restrict__ W,
    float* __restrict__ H)
{
    __shared__ __align__(32) _Float16 wfrag[4 * 8 * 512];
    const int tid = threadIdx.x;
    for (int idx = tid; idx < IN_F * OUT_F; idx += 256) {
        int k = idx >> 6, n = idx & 63;
        int t = n >> 4, s = k >> 5, koff = k & 31;
        wfrag[(((t * 8 + s) * 32) + frag_lane(n, koff)) * 16 + frag_e(koff)] =
            (_Float16)W[k * OUT_F + n];
    }
    __syncthreads();

    const int l = tid & 31, w = tid >> 5;
    const int i0 = (blockIdx.x * 8 + w) * 16;
    const int kb = (l >> 4) << 3;
    const float* row = X + (i0 + (l & 15)) * IN_F;

    v8f acc[4] = {};
#pragma unroll
    for (int s = 0; s < 8; ++s) {
        const float4* p0 = (const float4*)(row + s * 32 + kb);
        const float4* p1 = (const float4*)(row + s * 32 + 16 + kb);
        float4 f0 = p0[0], f1 = p0[1], f2 = p1[0], f3 = p1[1];
        v16h a;
        a[0]=(_Float16)f0.x; a[1]=(_Float16)f0.y; a[2]=(_Float16)f0.z; a[3]=(_Float16)f0.w;
        a[4]=(_Float16)f1.x; a[5]=(_Float16)f1.y; a[6]=(_Float16)f1.z; a[7]=(_Float16)f1.w;
        a[8]=(_Float16)f2.x; a[9]=(_Float16)f2.y; a[10]=(_Float16)f2.z; a[11]=(_Float16)f2.w;
        a[12]=(_Float16)f3.x; a[13]=(_Float16)f3.y; a[14]=(_Float16)f3.z; a[15]=(_Float16)f3.w;
#pragma unroll
        for (int t = 0; t < 4; ++t) {
            v16h b = *(const v16h*)&wfrag[(((t * 8 + s) * 32) + l) * 16];
            acc[t] = wmma_f16(a, b, acc[t]);
        }
    }
    const int mbase = i0 + ((l >> 4) << 3), n0 = l & 15;
#pragma unroll
    for (int t = 0; t < 4; ++t)
#pragma unroll
        for (int r = 0; r < 8; ++r)
            H[(mbase + r) * OUT_F + t * 16 + n0] = acc[t][r];
}

// ========== kernel 2: per-head dst scores, global max, E = exp ===========
__global__ __launch_bounds__(256) void k_head_exp(
    const float* __restrict__ H, const float* __restrict__ att,
    float* __restrict__ E)
{
    __shared__ float adst[OUT_F];
    __shared__ float red[256];
    const int h = blockIdx.x, tid = threadIdx.x;
    if (tid < OUT_F) adst[tid] = att[h * 2 * OUT_F + OUT_F + tid];
    __syncthreads();

    float d[NNODES / 256];
    float mx = -3.4e38f;
#pragma unroll 1
    for (int it = 0; it < NNODES / 256; ++it) {
        const float* hr = H + (tid + it * 256) * OUT_F;
        float s = 0.f;
#pragma unroll
        for (int f = 0; f < OUT_F; ++f) s += hr[f] * adst[f];
        d[it] = s;
        mx = fmaxf(mx, s);
    }
    red[tid] = mx;
    __syncthreads();
    for (int off = 128; off > 0; off >>= 1) {
        if (tid < off) red[tid] = fmaxf(red[tid], red[tid + off]);
        __syncthreads();
    }
    mx = red[0];
#pragma unroll 1
    for (int it = 0; it < NNODES / 256; ++it)
        E[h * NNODES + tid + it * 256] = expf(d[it] - mx);
}

// ==== kernel 3: build P in pre-swizzled WMMA B-fragment order (f16) ======
__global__ __launch_bounds__(256) void k_build_pfrag(
    const float* __restrict__ H, const float* __restrict__ E,
    _Float16* __restrict__ Pfrag)
{
    const int k = blockIdx.x;                 // node j = K index
    const int s = k >> 5, koff = k & 31;
    const int e = frag_e(koff);
    for (int n = threadIdx.x; n < NPAD; n += 256) {
        float v;
        if (n < 512)       v = E[(n >> 6) * NNODES + k] * H[k * OUT_F + (n & 63)];
        else if (n < 520)  v = E[(n - 512) * NNODES + k];
        else               v = 0.f;
        int t = n >> 4;
        Pfrag[(((t * KSTEPS + s) * 32) + frag_lane(n, koff)) * 16 + e] = (_Float16)v;
    }
}

// ==== kernel 3b: pre-convert adjacency -> f16 A-fragment layout ==========
// Afrag[mtile][kstep][lane][e]; pure bandwidth (64MB read, 32MB write).
__global__ __launch_bounds__(256) void k_build_afrag(
    const int* __restrict__ A, _Float16* __restrict__ Afrag)
{
    const int mtile = blockIdx.x;
    const int idx = threadIdx.x * 2;
    const int am = idx >> 5, koff = idx & 31;            // koff even
    const int ofs = frag_lane(am, koff) * 16 + frag_e(koff);
    const int* arow = A + (size_t)(mtile * 16 + am) * NNODES + koff;
    _Float16* dst = Afrag + (size_t)mtile * KSTEPS * 512 + ofs;
#pragma unroll 4
    for (int s = 0; s < KSTEPS; ++s) {
        int a0 = arow[s * 32], a1 = arow[s * 32 + 1];
        v2h pk; pk[0] = (_Float16)(float)a0; pk[1] = (_Float16)(float)a1;
        *(v2h*)(dst + (size_t)s * 512) = pk;
    }
}

// ====== kernel 4: T = A @ P, direct path (no LDS, 2-deep pipeline) =======
__global__ __launch_bounds__(256) void k_attn_gemm_direct(
    const _Float16* __restrict__ Afrag, const _Float16* __restrict__ Pfrag,
    float* __restrict__ T)
{
    const int tid = threadIdx.x;
    const int l = tid & 31, w = tid >> 5;
    const int mtile = blockIdx.x;
    const int start = w * 4;

    const v16h* ap  = (const v16h*)Afrag + (size_t)mtile * KSTEPS * 32 + l;
    const v16h* bp0 = (const v16h*)Pfrag + (size_t)(start + 0) * KSTEPS * 32 + l;
    const v16h* bp1 = (const v16h*)Pfrag + (size_t)(start + 1) * KSTEPS * 32 + l;
    const v16h* bp2 = (const v16h*)Pfrag + (size_t)(start + 2) * KSTEPS * 32 + l;
    const v16h* bp3 = (const v16h*)Pfrag + (size_t)(start + 3) * KSTEPS * 32 + l;
    const v16h* bp4 = (const v16h*)Pfrag + (size_t)32 * KSTEPS * 32 + l; // E cols

    v8f acc[4] = {};
    v8f acc4 = {};

    // pipeline prologue: register set 0 holds step s
    v16h a0 = ap[0];
    v16h c0 = bp0[0], c1 = bp1[0], c2 = bp2[0], c3 = bp3[0];

#pragma unroll 1
    for (int s = 0; s < KSTEPS; s += 2) {            // KSTEPS even
        const int s1 = s + 1;
        const int s2 = (s + 2 < KSTEPS) ? s + 2 : s + 1;

        // issue set-1 loads (step s+1) + wave0 E-column fragments up front
        v16h a1 = ap[s1 * 32];
        v16h d0 = bp0[s1 * 32], d1 = bp1[s1 * 32];
        v16h d2 = bp2[s1 * 32], d3 = bp3[s1 * 32];
        v16h e0, e1;
        if (w == 0) { e0 = bp4[s * 32]; e1 = bp4[s1 * 32]; }
        const int sp = (s + 8 < KSTEPS) ? s + 8 : s;
        __builtin_prefetch((const void*)(ap + sp * 32), 0, 1); // global_prefetch_b8

        // step s MACs on register set 0
        acc[0] = wmma_f16(a0, c0, acc[0]);
        acc[1] = wmma_f16(a0, c1, acc[1]);
        acc[2] = wmma_f16(a0, c2, acc[2]);
        acc[3] = wmma_f16(a0, c3, acc[3]);
        if (w == 0) acc4 = wmma_f16(a0, e0, acc4);   // wave-uniform branch

        // reload set 0 for step s+2 while set-1 MACs run
        a0 = ap[s2 * 32];
        c0 = bp0[s2 * 32]; c1 = bp1[s2 * 32];
        c2 = bp2[s2 * 32]; c3 = bp3[s2 * 32];

        // step s+1 MACs on register set 1
        acc[0] = wmma_f16(a1, d0, acc[0]);
        acc[1] = wmma_f16(a1, d1, acc[1]);
        acc[2] = wmma_f16(a1, d2, acc[2]);
        acc[3] = wmma_f16(a1, d3, acc[3]);
        if (w == 0) acc4 = wmma_f16(a1, e1, acc4);
    }

    const int mbase = mtile * 16 + ((l >> 4) << 3), n0c = l & 15;
#pragma unroll
    for (int tt = 0; tt < 4; ++tt)
#pragma unroll
        for (int r = 0; r < 8; ++r)
            T[(mbase + r) * NPAD + (start + tt) * 16 + n0c] = acc[tt][r];
    if (w == 0)
#pragma unroll
        for (int r = 0; r < 8; ++r)
            T[(mbase + r) * NPAD + 512 + n0c] = acc4[r];
}

// ====== kernel 4 fallback: LDS-staged A tile (small-workspace path) ======
__global__ __launch_bounds__(256) void k_attn_gemm_lds(
    const int* __restrict__ A, const _Float16* __restrict__ Pfrag,
    float* __restrict__ T)
{
    __shared__ __align__(32) _Float16 afrag[2][512];
    const int tid = threadIdx.x;
    const int l = tid & 31, w = tid >> 5;
    const int i0 = blockIdx.x * 16;
    const int start = w * 4;

    const int idx = tid * 2;
    const int am = idx >> 5, akoff = idx & 31;
    const int aofs = frag_lane(am, akoff) * 16 + frag_e(akoff);
    const int* arow = A + (size_t)(i0 + am) * NNODES + akoff;

    auto loadA = [&](int buf, int s) {
        int a0 = arow[s * 32], a1 = arow[s * 32 + 1];
        v2h pk; pk[0] = (_Float16)(float)a0; pk[1] = (_Float16)(float)a1;
        *(v2h*)&afrag[buf][aofs] = pk;
    };

    v8f acc[4] = {};
    v8f acc4 = {};

    loadA(0, 0);
    __syncthreads();
#pragma unroll 1
    for (int s = 0; s < KSTEPS; ++s) {
        const int cur = s & 1;
        if (s + 1 < KSTEPS) loadA(cur ^ 1, s + 1);
        v16h a = *(const v16h*)&afrag[cur][l * 16];
#pragma unroll
        for (int tt = 0; tt < 4; ++tt) {
            v16h b = *(const v16h*)(Pfrag +
                ((size_t)((start + tt) * KSTEPS + s) * 32 + l) * 16);
            acc[tt] = wmma_f16(a, b, acc[tt]);
        }
        if (w == 0) {
            v16h b = *(const v16h*)(Pfrag +
                ((size_t)(32 * KSTEPS + s) * 32 + l) * 16);
            acc4 = wmma_f16(a, b, acc4);
        }
        __syncthreads();
    }

    const int mbase = i0 + ((l >> 4) << 3), n0c = l & 15;
#pragma unroll
    for (int tt = 0; tt < 4; ++tt)
#pragma unroll
        for (int r = 0; r < 8; ++r)
            T[(mbase + r) * NPAD + (start + tt) * 16 + n0c] = acc[tt][r];
    if (w == 0)
#pragma unroll
        for (int r = 0; r < 8; ++r)
            T[(mbase + r) * NPAD + 512 + n0c] = acc4[r];
}

// ======= kernel 5: out[i,f] = (1/8) sum_h T[i,h*64+f] / T[i,512+h] =======
__global__ __launch_bounds__(256) void k_finalize(
    const float* __restrict__ T, float* __restrict__ out)
{
    const int gid = blockIdx.x * 256 + threadIdx.x;
    const int i = gid >> 6, f = gid & 63;
    const float* tr = T + i * NPAD;
    float s = 0.f;
#pragma unroll
    for (int h = 0; h < NHEADS; ++h)
        s += tr[h * 64 + f] / tr[512 + h];
    out[gid] = 0.125f * s;
}

// ================================ launch =================================
#define OFF_H  ((size_t)0)
#define OFF_E  ((size_t)(1u << 20))
#define OFF_P  (OFF_E + (1u << 17))
#define OFF_T  (OFF_P + (size_t)NTILES * KSTEPS * 512 * 2)
#define OFF_A  (OFF_T + (size_t)NNODES * NPAD * 4)
#define WS_BIG (OFF_A + (size_t)MTILES * KSTEPS * 512 * 2)   // ~45.5 MB

extern "C" void kernel_launch(void* const* d_in, const int* in_sizes, int n_in,
                              void* d_out, int out_size, void* d_ws, size_t ws_size,
                              hipStream_t stream) {
    const float* features = (const float*)d_in[0];   // (4096,256) f32
    const int*   adj      = (const int*)  d_in[1];   // (4096,4096) i32
    const float* W        = (const float*)d_in[2];   // (256,64) f32
    const float* att      = (const float*)d_in[3];   // (8,128,1) f32
    float* out = (float*)d_out;                      // (4096,64) f32

    char* ws = (char*)d_ws;
    float*    H     = (float*)(ws + OFF_H);
    float*    E     = (float*)(ws + OFF_E);
    _Float16* Pfrag = (_Float16*)(ws + OFF_P);
    float*    T     = (float*)(ws + OFF_T);
    _Float16* Afrag = (_Float16*)(ws + OFF_A);

    k_feat_gemm  <<<32,     256, 0, stream>>>(features, W, H);
    k_head_exp   <<<NHEADS, 256, 0, stream>>>(H, att, E);
    k_build_pfrag<<<NNODES, 256, 0, stream>>>(H, E, Pfrag);
    if (ws_size >= WS_BIG) {        // fixed per run -> deterministic
        k_build_afrag     <<<MTILES, 256, 0, stream>>>(adj, Afrag);
        k_attn_gemm_direct<<<MTILES, 256, 0, stream>>>(Afrag, Pfrag, T);
    } else {
        k_attn_gemm_lds   <<<MTILES, 256, 0, stream>>>(adj, Pfrag, T);
    }
    k_finalize<<<NNODES * OUT_F / 256, 256, 0, stream>>>(T, out);
}
